// Attention_71734543778365
// MI455X (gfx1250) — matrix operand order repaired and checked by reading, not compiled, not run on hardware
//
#include <hip/hip_runtime.h>

typedef __attribute__((ext_vector_type(16))) __bf16 v16bf;
typedef __attribute__((ext_vector_type(8)))  __bf16 v8bf;
typedef __attribute__((ext_vector_type(8)))  float  v8f;

#define NUM_HEADS      32
#define HEAD_DIM       128
#define NUM_KV_HEADS   8
#define SEQ_LEN        1024
#define BLOCK_SIZE     256
#define BLOCKS_PER_SEQ 4
#define ATTN_SCALE     0.08838834764831845f
#define KV_STRIDE      (NUM_KV_HEADS * HEAD_DIM)   /* floats between consecutive tokens */

__global__ __launch_bounds__(256)
void paged_attn_wmma_bf16(const float* __restrict__ query,
                          const float* __restrict__ k_cache,
                          const float* __restrict__ v_cache,
                          const int*   __restrict__ block_tables,
                          float*       __restrict__ out)
{
    // per-wave, per-subtile P staging: C-layout -> A-layout transpose through LDS
    // (in-wave LDS ops are processed in order on CDNA5, so no barrier needed)
    __shared__ __bf16 Psh[8][2][16][32];

    const int lane = threadIdx.x & 31;
    const int wave = threadIdx.x >> 5;
    const int ln15 = lane & 15;     // column / key / q-row selector inside a 16-lane group
    const int hi   = lane >> 4;     // which 16-lane half of the wave

    const int gw = blockIdx.x * 8 + wave;    // 8192 waves total
    const int qt = gw & 31;                  // 32 q-tiles of 32 rows per (b,h)
    const int h  = (gw >> 5) & 31;
    const int b  = gw >> 10;
    const int q0 = qt << 5;                  // 32 q-rows per wave (2 x 16-row subtiles)
    const int hkv = h >> 2;                  // consecutive GQA repeat: head h -> kv head h/4

    // ---- resident Q tiles: 2 subtiles x 4 A-fragments (16x32 bf16), ISA wave32 A layout ----
    v16bf Qf[2][4];
#pragma unroll
    for (int t = 0; t < 2; ++t) {
        const float* qp = query +
            ((size_t)((b * SEQ_LEN + q0 + 16 * t + ln15) * NUM_HEADS + h)) * HEAD_DIM;
#pragma unroll
        for (int c = 0; c < 4; ++c) {
            const int f0 = 32 * c + hi * 8;
#pragma unroll
            for (int e = 0; e < 8; ++e) Qf[t][c][e]     = (__bf16)qp[f0 + e];
#pragma unroll
            for (int e = 0; e < 8; ++e) Qf[t][c][8 + e] = (__bf16)qp[f0 + 16 + e];
        }
    }

    const v8f vzero = {};
    v8f   o[2][8];
    float m[2][8], l[2][8];
#pragma unroll
    for (int t = 0; t < 2; ++t) {
#pragma unroll
        for (int n = 0; n < 8; ++n) o[t][n] = vzero;
#pragma unroll
        for (int j = 0; j < 8; ++j) { m[t][j] = -3.0e38f; l[t][j] = 0.0f; }
    }

    // 32-key blocks with k0 <= q0+31; since k0,q0 are multiples of 32 every processed
    // block has k0 <= q0, so each row always sees >=1 unmasked key (no -inf corner cases)
    const int nkb = (q0 >> 5) + 1;

    for (int kb = 0; kb < nkb; ++kb) {
        const int    k0      = kb << 5;
        const int    pb      = block_tables[b * BLOCKS_PER_SEQ + (k0 >> 8)];
        const size_t tokbase = ((size_t)pb * BLOCK_SIZE + (k0 & 255)) * NUM_KV_HEADS + hkv;
        const float* kc      = k_cache + tokbase * HEAD_DIM;
        const float* vc      = v_cache + tokbase * HEAD_DIM;

        if (kb + 1 < nkb) {     // gfx1250 global_prefetch_b8 of next K block
            const int    k0n = (kb + 1) << 5;
            const int    pbn = block_tables[b * BLOCKS_PER_SEQ + (k0n >> 8)];
            const float* kn  = k_cache +
                (((size_t)pbn * BLOCK_SIZE + (k0n & 255)) * NUM_KV_HEADS + hkv) * HEAD_DIM;
            __builtin_prefetch(kn + (size_t)ln15 * KV_STRIDE, 0, 0);
        }

        // ---- scores: each K B-fragment feeds BOTH q-subtiles (2x WMMA reuse) ----
        v8f s[2][2];            // [subtile][key-half]
        s[0][0] = vzero; s[0][1] = vzero; s[1][0] = vzero; s[1][1] = vzero;
#pragma unroll
        for (int half = 0; half < 2; ++half) {
            const float* kp = kc + (size_t)(16 * half + ln15) * KV_STRIDE;
#pragma unroll
            for (int c = 0; c < 4; ++c) {
                v16bf Bf;       // B 32x16: lane group h holds K 8h..8h+7, 16+8h..16+8h+7
                const int f0 = 32 * c + hi * 8;
#pragma unroll
                for (int e = 0; e < 16; ++e) Bf[e] = (__bf16)kp[f0 + e + (e >= 8 ? 8 : 0)];
                s[0][half] = __builtin_amdgcn_wmma_f32_16x16x32_bf16(
                    false, Qf[0][c], false, Bf, (short)0, s[0][half], false, false);
                s[1][half] = __builtin_amdgcn_wmma_f32_16x16x32_bf16(
                    false, Qf[1][c], false, Bf, (short)0, s[1][half], false, false);
            }
        }

        // ---- online softmax per subtile (per C-layout row j) ----
#pragma unroll
        for (int t = 0; t < 2; ++t) {
#pragma unroll
            for (int j = 0; j < 8; ++j) {
                const int qrow = q0 + 16 * t + j + hi * 8;
                float sc0 = s[t][0][j] * ATTN_SCALE;
                float sc1 = s[t][1][j] * ATTN_SCALE;
                if (k0 + ln15      > qrow) sc0 = -3.0e38f;   // causal mask
                if (k0 + 16 + ln15 > qrow) sc1 = -3.0e38f;
                float mb = fmaxf(sc0, sc1);
#pragma unroll
                for (int off = 8; off >= 1; off >>= 1)
                    mb = fmaxf(mb, __shfl_xor(mb, off, 16));
                const float mnew = fmaxf(m[t][j], mb);
                const float corr = __expf(m[t][j] - mnew);
                const float p0   = __expf(sc0 - mnew);
                const float p1   = __expf(sc1 - mnew);
                float ps = p0 + p1;
#pragma unroll
                for (int off = 8; off >= 1; off >>= 1)
                    ps += __shfl_xor(ps, off, 16);
                l[t][j] = l[t][j] * corr + ps;
                m[t][j] = mnew;
#pragma unroll
                for (int n = 0; n < 8; ++n) o[t][n][j] *= corr;
                Psh[wave][t][j + hi * 8][ln15]      = (__bf16)p0;
                Psh[wave][t][j + hi * 8][16 + ln15] = (__bf16)p1;
            }
        }

        // ---- P (16x32) A-fragments: two aligned 16B LDS vector loads each ----
        v16bf Pf[2];
#pragma unroll
        for (int t = 0; t < 2; ++t) {
            const __bf16* prow = &Psh[wave][t][ln15][0];
            const v8bf plo = *(const v8bf*)(prow + hi * 8);        // K = hi*8 .. hi*8+7
            const v8bf phi = *(const v8bf*)(prow + 16 + hi * 8);   // K = 16+hi*8 ..
            Pf[t] = __builtin_shufflevector(plo, phi,
                                            0, 1, 2, 3, 4, 5, 6, 7,
                                            8, 9, 10, 11, 12, 13, 14, 15);
        }

        // ---- O(32x128) += P * V: each V B-fragment feeds BOTH subtiles ----
#pragma unroll
        for (int n = 0; n < 8; ++n) {
            v16bf Vf;
#pragma unroll
            for (int e = 0; e < 16; ++e) {
                const int kk = hi * 8 + e + (e >= 8 ? 8 : 0);
                Vf[e] = (__bf16)vc[(size_t)kk * KV_STRIDE + 16 * n + ln15];
            }
            o[0][n] = __builtin_amdgcn_wmma_f32_16x16x32_bf16(
                false, Pf[0], false, Vf, (short)0, o[0][n], false, false);
            o[1][n] = __builtin_amdgcn_wmma_f32_16x16x32_bf16(
                false, Pf[1], false, Vf, (short)0, o[1][n], false, false);
        }
    }

    // ---- normalize and store (fp32 output, same layout as query) ----
#pragma unroll
    for (int t = 0; t < 2; ++t) {
#pragma unroll
        for (int j = 0; j < 8; ++j) {
            const float inv = 1.0f / l[t][j];
            float* op = out +
                ((size_t)((b * SEQ_LEN + q0 + 16 * t + j + hi * 8) * NUM_HEADS + h)) * HEAD_DIM
                + ln15;
#pragma unroll
            for (int n = 0; n < 8; ++n) op[16 * n] = o[t][n][j] * inv;
        }
    }
}

extern "C" void kernel_launch(void* const* d_in, const int* in_sizes, int n_in,
                              void* d_out, int out_size, void* d_ws, size_t ws_size,
                              hipStream_t stream) {
    const float* q  = (const float*)d_in[0];
    const float* k  = (const float*)d_in[1];
    const float* v  = (const float*)d_in[2];
    const int*   bt = (const int*)d_in[3];
    float*       o  = (float*)d_out;

    // 8 seqs * 32 heads * 32 q-tiles(32 rows) = 8192 waves / 8 waves per block
    dim3 grid(1024), block(256);
    hipLaunchKernelGGL(paged_attn_wmma_bf16, grid, block, 0, stream, q, k, v, bt, o);
}
